// VectorQuantizer_6253472383384
// MI455X (gfx1250) — matrix operand order repaired and checked by reading, not compiled, not run on hardware
//
#include <hip/hip_runtime.h>
#include <hip/hip_bf16.h>
#include <math.h>

typedef __bf16 bf16_t;
typedef __attribute__((ext_vector_type(16))) __bf16 v16bf;
typedef __attribute__((ext_vector_type(8)))  float  v8f;

#define N_ROWS   32768
#define DIM      256
#define KCODES   1024
#define BLOCK_M  32
#define DSTRIDE  1028      // pad vs 1024: 8*1028 % 64 == 32 -> no LDS bank conflict between lane halves
#define INV_T    100.0f    // 1 / ENTROPY_TEMPERATURE

// ---------------- prep: codebook -> bf16 hi/lo planes + norms, zero accumulators ----------------
__global__ __launch_bounds__(256) void vq_prep(const float* __restrict__ cb,
                                               bf16_t* __restrict__ cb_hi,
                                               bf16_t* __restrict__ cb_lo,
                                               float* __restrict__ cnorm,
                                               float* __restrict__ colsum,
                                               double* __restrict__ scal) {
    __shared__ float sp[256];
    int k = blockIdx.x, t = threadIdx.x;
    float v = cb[k * DIM + t];
    bf16_t hb = (bf16_t)v;
    cb_hi[k * DIM + t] = hb;
    cb_lo[k * DIM + t] = (bf16_t)(v - (float)hb);
    sp[t] = v * v;
    __syncthreads();
    for (int s = 128; s > 0; s >>= 1) { if (t < s) sp[t] += sp[t + s]; __syncthreads(); }
    if (t == 0) cnorm[k] = sp[0];
    if (k < 4) colsum[k * 256 + t] = 0.f;            // zero avg_probs accumulator (1024 floats)
    if (k == 0 && t < 2) scal[t] = 0.0;              // scal[0]=sum(q-x)^2, scal[1]=sum_n sum_k p*logp
}

// ---------------- main: distances via bf16x3 WMMA, argmin, softmax stats, quantized out ----------------
__global__ __launch_bounds__(256) void vq_main(const float* __restrict__ x,
                                               const float* __restrict__ cb,
                                               const bf16_t* __restrict__ cb_hi,
                                               const bf16_t* __restrict__ cb_lo,
                                               const float* __restrict__ cnorm,
                                               float* __restrict__ colsum,
                                               double* __restrict__ scal,
                                               float* __restrict__ out_q,
                                               float* __restrict__ out_idx) {
    extern __shared__ char smem[];
    bf16_t* sAh = (bf16_t*)smem;                                  // 32*256 bf16 = 16384 B
    bf16_t* sAl = (bf16_t*)(smem + 16384);                        // 16384 B
    float*  sD  = (float*)(smem + 32768);                         // 32*1028 f32 = 131584 B
    float*  sXn = (float*)(smem + 32768 + 131584);                // 32 f32
    float*  sRm = sXn + BLOCK_M;                                  // 32 f32 (row min dist)
    float*  sRz = sRm + BLOCK_M;                                  // 32 f32 (1/Z)

    const int t    = threadIdx.x;
    const int lane = t & 31;
    const int w    = t >> 5;
    const int r0   = blockIdx.x * BLOCK_M;

    // ---- stage 1: load x tile, split into bf16 hi/lo, compute |x|^2 per row ----
    {
        int r  = t >> 3;                 // 0..31
        int c0 = (t & 7) * 32;           // 0..224
        const float4* xr = (const float4*)(x + (size_t)(r0 + r) * DIM + c0);
        float s = 0.f;
        #pragma unroll
        for (int i = 0; i < 8; ++i) {
            float4 v4 = xr[i];
            float vv[4] = {v4.x, v4.y, v4.z, v4.w};
            #pragma unroll
            for (int jj = 0; jj < 4; ++jj) {
                float v = vv[jj];
                s += v * v;
                bf16_t hb = (bf16_t)v;
                sAh[r * DIM + c0 + i * 4 + jj] = hb;
                sAl[r * DIM + c0 + i * 4 + jj] = (bf16_t)(v - (float)hb);
            }
        }
        s += __shfl_xor(s, 1); s += __shfl_xor(s, 2); s += __shfl_xor(s, 4);
        if ((t & 7) == 0) sXn[r] = s;
    }
    __syncthreads();

    // ---- stage 2: GEMM -2*x.cT + norms -> distances in LDS ----
    // wave w: row tile mt = w&1 (16 rows), column tiles ct0..ct0+15 (16 cols each)
    const int h    = lane >> 4;      // lane half
    const int nl   = lane & 15;
    const int mrow = (w & 1) << 4;
    const int ct0  = (w >> 1) << 4;
    const int mA   = mrow + nl;      // A-fragment row for this lane

    // Preload A fragments (16x32 bf16 layout, ISA 7.12.2):
    //   lane half h: VGPR0-3 = K in [8h, 8h+8), VGPR4-7 = K in [16+8h, 16+8h+8)
    v16bf ahi[8], alo[8];
    #pragma unroll
    for (int dc = 0; dc < 8; ++dc) {
        int d0 = dc * 32;
        const bf16_t* ph = &sAh[mA * DIM + d0 + 8 * h];
        const bf16_t* pl = &sAl[mA * DIM + d0 + 8 * h];
        ((float4*)&ahi[dc])[0] = *(const float4*)(ph);
        ((float4*)&ahi[dc])[1] = *(const float4*)(ph + 16);
        ((float4*)&alo[dc])[0] = *(const float4*)(pl);
        ((float4*)&alo[dc])[1] = *(const float4*)(pl + 16);
    }
    float xnv[8];
    #pragma unroll
    for (int v = 0; v < 8; ++v) xnv[v] = sXn[mrow + v + 8 * h];

    for (int ti = 0; ti < 16; ++ti) {
        int ct = ct0 + ti;
        int n  = (ct << 4) + nl;     // codebook row = B column
        // B fragment (32x16 bf16): lane half h holds K in [8h, 8h+8) and [16+8h, 16+8h+8) of column n (matches A)
        const bf16_t* bhp = cb_hi + (size_t)n * DIM + 8 * h;
        const bf16_t* blp = cb_lo + (size_t)n * DIM + 8 * h;
        float cn = cnorm[n];
        v8f acc = {0.f, 0.f, 0.f, 0.f, 0.f, 0.f, 0.f, 0.f};
        #pragma unroll
        for (int dc = 0; dc < 8; ++dc) {
            v16bf bhv, blv;
            ((float4*)&bhv)[0] = *(const float4*)(bhp + dc * 32);
            ((float4*)&bhv)[1] = *(const float4*)(bhp + dc * 32 + 16);
            ((float4*)&blv)[0] = *(const float4*)(blp + dc * 32);
            ((float4*)&blv)[1] = *(const float4*)(blp + dc * 32 + 16);
            // bf16x3 split-precision accumulate: hi*hi + hi*lo + lo*hi
            acc = __builtin_amdgcn_wmma_f32_16x16x32_bf16(false, ahi[dc], false, bhv, (short)0, acc, false, false);
            acc = __builtin_amdgcn_wmma_f32_16x16x32_bf16(false, ahi[dc], false, blv, (short)0, acc, false, false);
            acc = __builtin_amdgcn_wmma_f32_16x16x32_bf16(false, alo[dc], false, bhv, (short)0, acc, false, false);
        }
        #pragma unroll
        for (int v = 0; v < 8; ++v) {
            int m = mrow + v + 8 * h;    // C/D layout: VGPR v -> rows v / v+8 by lane half
            sD[m * DSTRIDE + (ct << 4) + nl] = xnv[v] - 2.0f * acc[v] + cn;
        }
    }
    __syncthreads();

    // ---- stage 3: per-row argmin + softmax stats + quantized output ----
    #pragma unroll 1
    for (int j = 0; j < 4; ++j) {
        int r  = (w << 2) + j;
        int gr = r0 + r;
        const float* dr = sD + r * DSTRIDE;

        float mv = 3.402823466e+38f; int mi = 0;
        for (int c = lane; c < KCODES; c += 32) {
            float dv = dr[c];
            if (dv < mv) { mv = dv; mi = c; }
        }
        #pragma unroll
        for (int off = 16; off > 0; off >>= 1) {
            float ov = __shfl_xor(mv, off);
            int   oi = __shfl_xor(mi, off);
            if (ov < mv || (ov == mv && oi < mi)) { mv = ov; mi = oi; }  // first-occurrence tie-break
        }

        float Z = 0.f, U = 0.f;
        for (int c = lane; c < KCODES; c += 32) {
            float u = (mv - dr[c]) * INV_T;
            float e = __expf(u);
            Z += e; U += u * e;
        }
        #pragma unroll
        for (int off = 16; off > 0; off >>= 1) { Z += __shfl_xor(Z, off); U += __shfl_xor(U, off); }

        if (lane == 0) {
            out_idx[gr] = (float)mi;
            atomicAdd(&scal[1], (double)(U / Z - logf(Z)));   // sum_k p*logp for this row
            sRm[r] = mv; sRz[r] = 1.0f / Z;
        }

        const float* crow = cb + (size_t)mi * DIM;
        const float* xrow = x + (size_t)gr * DIM;
        float*       qrow = out_q + (size_t)gr * DIM;
        float ms = 0.f;
        for (int c = lane; c < DIM; c += 32) {
            float q = crow[c], xv = xrow[c];
            qrow[c] = q;                        // straight-through: forward value == quantized
            float d = q - xv; ms += d * d;
        }
        #pragma unroll
        for (int off = 16; off > 0; off >>= 1) ms += __shfl_xor(ms, off);
        if (lane == 0) atomicAdd(&scal[0], (double)ms);
    }
    __syncthreads();

    // ---- stage 4: column sums of probabilities (avg_probs numerator) ----
    for (int c = t; c < KCODES; c += 256) {
        float s = 0.f;
        #pragma unroll 4
        for (int r = 0; r < BLOCK_M; ++r)
            s += __expf((sRm[r] - sD[r * DSTRIDE + c]) * INV_T) * sRz[r];
        atomicAdd(&colsum[c], s);
    }
}

// ---------------- finalize: assemble scalar loss ----------------
__global__ __launch_bounds__(256) void vq_finalize(const float* __restrict__ colsum,
                                                   const double* __restrict__ scal,
                                                   float* __restrict__ out_loss) {
    __shared__ float sp[256];
    int t = threadIdx.x;
    float local = 0.f;
    for (int c = t; c < KCODES; c += 256) {
        float ap = colsum[c] * (1.0f / (float)N_ROWS);
        local += ap * logf(ap + 1e-5f);
    }
    sp[t] = local;
    __syncthreads();
    for (int s = 128; s > 0; s >>= 1) { if (t < s) sp[t] += sp[t + s]; __syncthreads(); }
    if (t == 0) {
        float avg_entropy    = -sp[0];
        float sample_entropy = -(float)(scal[1] / (double)N_ROWS);
        float mse_mean       =  (float)(scal[0] / ((double)N_ROWS * (double)DIM));
        float ent            = sample_entropy - avg_entropy;
        *out_loss = 1.25f * mse_mean + 0.1f * ent;   // (0.25 + 1.0)*mse + 0.1*entropy_loss
    }
}

extern "C" void kernel_launch(void* const* d_in, const int* in_sizes, int n_in,
                              void* d_out, int out_size, void* d_ws, size_t ws_size,
                              hipStream_t stream) {
    (void)in_sizes; (void)n_in; (void)out_size; (void)ws_size;
    const float* x  = (const float*)d_in[0];
    const float* cb = (const float*)d_in[1];
    float* out = (float*)d_out;

    char* ws = (char*)d_ws;
    bf16_t* cb_hi  = (bf16_t*)ws;                    // 524288 B
    bf16_t* cb_lo  = (bf16_t*)(ws + 524288);         // 524288 B
    float*  cnorm  = (float*)(ws + 1048576);         // 4096 B
    float*  colsum = (float*)(ws + 1052672);         // 4096 B
    double* scal   = (double*)(ws + 1056768);        // 16 B

    float* out_q    = out;                           // 32768*256 floats
    float* out_loss = out + 8388608;                 // 1 float
    float* out_idx  = out + 8388609;                 // 32768 floats (indices as float)

    vq_prep<<<KCODES, 256, 0, stream>>>(cb, cb_hi, cb_lo, cnorm, colsum, scal);

    size_t smem = 32768u + (size_t)BLOCK_M * DSTRIDE * 4u + 3u * BLOCK_M * 4u;  // 164736 B
    hipFuncSetAttribute(reinterpret_cast<const void*>(vq_main),
                        hipFuncAttributeMaxDynamicSharedMemorySize, (int)smem);
    vq_main<<<N_ROWS / BLOCK_M, 256, smem, stream>>>(x, cb, cb_hi, cb_lo, cnorm,
                                                     colsum, scal, out_q, out_idx);

    vq_finalize<<<1, 256, 0, stream>>>(colsum, scal, out_loss);
}
